// ContextAwareDeepGraphEncoder_20976620273701
// MI455X (gfx1250) — matrix operand order repaired and checked by reading, not compiled, not run on hardware
//
#include <hip/hip_runtime.h>
#include <hip/hip_bf16.h>

// ---------------------------------------------------------------------------
// Problem constants (match reference)
// ---------------------------------------------------------------------------
#define NN    50000        // nodes
#define EE    800000       // edges
#define FDIM  256          // feature dim everywhere (F_IN == H == OUT)
#define N2    (2 * NN)     // concat node count
#define E3    (EE + N2)    // edges + self loops = 900000
#define EPSV  1e-5f
#define NSLP  0.2f

typedef __attribute__((ext_vector_type(16))) __bf16 v16bf;
typedef __attribute__((ext_vector_type(8)))  __bf16 v8bf;
typedef __attribute__((ext_vector_type(8)))  float  v8f;

// ---------------------------------------------------------------------------
// Helpers
// ---------------------------------------------------------------------------
__device__ __forceinline__ float wave_reduce_sum(float v) {
#pragma unroll
    for (int off = 16; off > 0; off >>= 1) v += __shfl_xor(v, off, 32);
    return v;
}

// monotonic float <-> ordered-uint mapping for atomicMax on floats
__device__ __forceinline__ unsigned ford(float f) {
    unsigned u = __float_as_uint(f);
    return (u & 0x80000000u) ? ~u : (u | 0x80000000u);
}
__device__ __forceinline__ float dorf(unsigned u) {
    return (u & 0x80000000u) ? __uint_as_float(u & 0x7fffffffu)
                             : __uint_as_float(~u);
}

// A fragment from row-major bf16 [rows][256].
// ISA 16-bit A layout: lane L (group g=L>>4) holds row m0+(L&15),
// K = k0 + {g*8 .. g*8+7} and k0+16+{g*8 .. g*8+7}  -> two contiguous 16B chunks.
__device__ __forceinline__ v16bf load_frag_a(const __bf16* __restrict__ base,
                                             int row, int k0, int lane) {
    const int kb = (lane >> 4) * 8;
    const __bf16* p = base + (size_t)row * FDIM + k0 + kb;
    union { v16bf v; v8bf h[2]; } u;
    u.h[0] = *(const v8bf*)(p);
    u.h[1] = *(const v8bf*)(p + 16);
    return u.v;
}

// B fragment from TRANSPOSED bf16 weight Wt[col][k] (col-major original).
// Column-major B layout: lane = column, group g=lane>>4 holds
// K = k0 + {g*8 .. g*8+7} and k0+16+{g*8 .. g*8+7}  -> two contiguous 16B chunks.
__device__ __forceinline__ v16bf load_frag_b(const __bf16* __restrict__ Wt,
                                             int col, int k0, int lane) {
    const int kb = (lane >> 4) * 8;
    const __bf16* p = Wt + (size_t)col * FDIM + k0 + kb;
    union { v16bf v; v8bf h[2]; } u;
    u.h[0] = *(const v8bf*)(p);
    u.h[1] = *(const v8bf*)(p + 16);
    return u.v;
}

// ---------------------------------------------------------------------------
// Init / packing kernels
// ---------------------------------------------------------------------------
__global__ void fillf_kernel(float* p, int n, float v) {
    int i = blockIdx.x * 256 + threadIdx.x;
    if (i < n) p[i] = v;
}
__global__ void fillu_kernel(unsigned* p, int n, unsigned v) {
    int i = blockIdx.x * 256 + threadIdx.x;
    if (i < n) p[i] = v;
}
// out[0:2N*256) = bias broadcast; out[2N*256] = ss_loss = 0
__global__ void out_init_kernel(float* __restrict__ out,
                                const float* __restrict__ bias) {
    int i = blockIdx.x * 256 + threadIdx.x;           // exact 2N*256 threads
    out[i] = bias[i & (FDIM - 1)];
    if (i == 0) out[(size_t)N2 * FDIM] = 0.0f;
}

// f32 -> bf16 row-major copy
__global__ void packbf_kernel(const float* __restrict__ src,
                              __bf16* __restrict__ dst, int n) {
    int i = blockIdx.x * 256 + threadIdx.x;
    if (i < n) dst[i] = (__bf16)src[i];
}

// transpose-pack weight: W[k][c] f32 (256x256) -> Wt[c][k] bf16
__global__ void packWt_kernel(const float* __restrict__ W,
                              __bf16* __restrict__ Wt) {
    int i = blockIdx.x * 256 + threadIdx.x;   // exact 65536 threads
    int k = i >> 8, c = i & 255;
    Wt[(size_t)c * FDIM + k] = (__bf16)W[i];
}

// agg f32 -> agg_bf (sum) and mean_bf (sum * invdeg[node]) bf16
__global__ void pack_agg_kernel(const float* __restrict__ agg,
                                const float* __restrict__ invdeg,
                                __bf16* __restrict__ agg_bf,
                                __bf16* __restrict__ mean_bf) {
    int i = blockIdx.x * 256 + threadIdx.x;   // exact N*256 threads
    float a = agg[i];
    agg_bf[i]  = (__bf16)a;
    mean_bf[i] = (__bf16)(a * invdeg[i >> 8]);
}

// ---------------------------------------------------------------------------
// 1) neighbor aggregation: agg[dst] += x[src], deg[dst] += 1
//    one wave per edge, 8 floats per lane
// ---------------------------------------------------------------------------
__global__ void scatter_agg_kernel(const int* __restrict__ src,
                                   const int* __restrict__ dst,
                                   const float* __restrict__ x,
                                   float* __restrict__ agg,
                                   float* __restrict__ deg) {
    const int e    = (blockIdx.x * 256 + threadIdx.x) >> 5;   // exact E waves
    const int lane = threadIdx.x & 31;
    const int s = src[e], d = dst[e];
    const float* xs = x   + (size_t)s * FDIM + lane * 8;
    float*       ad = agg + (size_t)d * FDIM + lane * 8;
#pragma unroll
    for (int i = 0; i < 8; ++i) atomicAdd(&ad[i], xs[i]);
    if (lane == 0) atomicAdd(&deg[d], 1.0f);
}

__global__ void invdeg_kernel(const float* __restrict__ deg,
                              float* __restrict__ invdeg) {
    int i = blockIdx.x * 256 + threadIdx.x;
    if (i < NN) invdeg[i] = 1.0f / fmaxf(deg[i], 1.0f);
}

// ---------------------------------------------------------------------------
// 2) fused layer-1/2 GEMMs (WMMA bf16, f32 accumulate), all operands
//    pre-packed bf16 so the hot loop is pure b128 loads + v_wmma.
//    h1 = mean@Wl + bl + x@Wr      -> xc[0:N)
//    h2 = agg @Wrel + brel + x@Wroot -> xc[N:2N)
//    one wave per 16x16 output tile; 50000 tiles, 8 waves/block
// ---------------------------------------------------------------------------
__global__ void gemm_h_kernel(const __bf16* __restrict__ x_bf,
                              const __bf16* __restrict__ agg_bf,
                              const __bf16* __restrict__ mean_bf,
                              const __bf16* __restrict__ WlT,
                              const __bf16* __restrict__ WrT,
                              const __bf16* __restrict__ WrelT,
                              const __bf16* __restrict__ WrootT,
                              const float* __restrict__ bl,
                              const float* __restrict__ brel,
                              float* __restrict__ xc) {
    const int wave = (blockIdx.x << 3) + (threadIdx.x >> 5);
    const int lane = threadIdx.x & 31;
    const int m0 = (wave >> 4) * 16;          // row tile (3125 tiles)
    const int n0 = (wave & 15) * 16;          // col tile (16 tiles)
    const int arow = m0 + (lane & 15);
    const int bcol = n0 + (lane & 15);

    v8f acc1 = {}; v8f acc2 = {};
#pragma unroll
    for (int k0 = 0; k0 < FDIM; k0 += 32) {
        v16bf ax = load_frag_a(x_bf,    arow, k0, lane);
        v16bf as = load_frag_a(agg_bf,  arow, k0, lane);
        v16bf am = load_frag_a(mean_bf, arow, k0, lane);
        v16bf bWl    = load_frag_b(WlT,    bcol, k0, lane);
        v16bf bWr    = load_frag_b(WrT,    bcol, k0, lane);
        v16bf bWrel  = load_frag_b(WrelT,  bcol, k0, lane);
        v16bf bWroot = load_frag_b(WrootT, bcol, k0, lane);
        acc1 = __builtin_amdgcn_wmma_f32_16x16x32_bf16(false, am, false, bWl,
                                                       (short)0, acc1, false, false);
        acc1 = __builtin_amdgcn_wmma_f32_16x16x32_bf16(false, ax, false, bWr,
                                                       (short)0, acc1, false, false);
        acc2 = __builtin_amdgcn_wmma_f32_16x16x32_bf16(false, as, false, bWrel,
                                                       (short)0, acc2, false, false);
        acc2 = __builtin_amdgcn_wmma_f32_16x16x32_bf16(false, ax, false, bWroot,
                                                       (short)0, acc2, false, false);
    }
    const float b1 = bl[bcol], b2 = brel[bcol];
    const int rbase = m0 + ((lane >> 4) << 3);   // C/D layout: lanes 16-31 hold M=r+8
#pragma unroll
    for (int r = 0; r < 8; ++r) {
        xc[(size_t)(rbase + r) * FDIM + bcol]      = acc1[r] + b1;
        xc[(size_t)(NN + rbase + r) * FDIM + bcol] = acc2[r] + b2;
    }
}

// ---------------------------------------------------------------------------
// 3) GraphNorm column statistics (sum, sumsq for h1 and h2)
// ---------------------------------------------------------------------------
__global__ void colstats_kernel(const float* __restrict__ xc,
                                float* __restrict__ stats,
                                int rows_per_block) {
    const int c  = threadIdx.x;
    const int r0 = blockIdx.x * rows_per_block;
    int r1 = r0 + rows_per_block; if (r1 > NN) r1 = NN;
    float s1 = 0.f, q1 = 0.f, s2 = 0.f, q2 = 0.f;
    for (int r = r0; r < r1; ++r) {
        float a = xc[(size_t)r * FDIM + c];
        float b = xc[(size_t)(NN + r) * FDIM + c];
        s1 += a; q1 += a * a;
        s2 += b; q2 += b * b;
    }
    atomicAdd(&stats[c], s1);
    atomicAdd(&stats[FDIM + c], q1);
    atomicAdd(&stats[2 * FDIM + c], s2);
    atomicAdd(&stats[3 * FDIM + c], q2);
}

// scale/shift per column:  out = h*scale + shift ; shift = beta - scale*alpha*mean
__global__ void scaleshift_kernel(const float* __restrict__ stats,
                                  const float* __restrict__ gamma,
                                  const float* __restrict__ beta,
                                  const float* __restrict__ alpha,
                                  float* __restrict__ ss) {
    const int c = threadIdx.x;
    const float inv = 1.0f / (float)NN;
    const float a = alpha[c], g = gamma[c], bt = beta[c];
#pragma unroll
    for (int half = 0; half < 2; ++half) {
        float m  = stats[2 * FDIM * half + c] * inv;
        float q  = stats[2 * FDIM * half + FDIM + c] * inv;
        float var = q - 2.0f * a * m * m + a * a * m * m;   // E[(h-alpha*m)^2]
        float sc = g * rsqrtf(var + EPSV);
        ss[2 * FDIM * half + c]        = sc;
        ss[2 * FDIM * half + FDIM + c] = bt - sc * a * m;
    }
}

// normalize + ReLU; emit bf16 activations for the next WMMA GEMM
__global__ void norm_relu_kernel(const float* __restrict__ xc,
                                 const float* __restrict__ ss,
                                 __bf16* __restrict__ xcb) {
    const int i = blockIdx.x * 256 + threadIdx.x;     // exact 2N*256 threads
    const int c = i & (FDIM - 1);
    const int half = (i >= NN * FDIM) ? 1 : 0;
    const float sc = ss[2 * FDIM * half + c];
    const float sh = ss[2 * FDIM * half + FDIM + c];
    float v = xc[i] * sc + sh;
    xcb[i] = (__bf16)(v > 0.0f ? v : 0.0f);
}

// ---------------------------------------------------------------------------
// 4) xp = relu(norm(h)) @ sg_W   (WMMA, 100000 tiles)
// ---------------------------------------------------------------------------
__global__ void gemm_xp_kernel(const __bf16* __restrict__ xcb,
                               const __bf16* __restrict__ WT,
                               float* __restrict__ xp) {
    const int wave = (blockIdx.x << 3) + (threadIdx.x >> 5);
    const int lane = threadIdx.x & 31;
    const int m0 = (wave >> 4) * 16;          // 6250 row tiles
    const int n0 = (wave & 15) * 16;
    const int arow = m0 + (lane & 15);
    const int bcol = n0 + (lane & 15);

    v8f acc = {};
#pragma unroll
    for (int k0 = 0; k0 < FDIM; k0 += 32) {
        v16bf a = load_frag_a(xcb, arow, k0, lane);
        v16bf b = load_frag_b(WT, bcol, k0, lane);
        acc = __builtin_amdgcn_wmma_f32_16x16x32_bf16(false, a, false, b,
                                                      (short)0, acc, false, false);
    }
    const int rbase = m0 + ((lane >> 4) << 3);
#pragma unroll
    for (int r = 0; r < 8; ++r)
        xp[(size_t)(rbase + r) * FDIM + bcol] = acc[r];
}

// ---------------------------------------------------------------------------
// 5) per-node GO logits: gl = xp@att_l, gr = xp@att_r   (one wave per node)
// ---------------------------------------------------------------------------
__global__ void node_att_kernel(const float* __restrict__ xp,
                                const float* __restrict__ att_l,
                                const float* __restrict__ att_r,
                                float* __restrict__ gl,
                                float* __restrict__ gr) {
    const int n    = (blockIdx.x * 256 + threadIdx.x) >> 5;   // exact 2N waves
    const int lane = threadIdx.x & 31;
    const float* row = xp + (size_t)n * FDIM + lane * 8;
    const float* al  = att_l + lane * 8;
    const float* ar  = att_r + lane * 8;
    float sl = 0.f, sr = 0.f;
#pragma unroll
    for (int i = 0; i < 8; ++i) { float v = row[i]; sl += v * al[i]; sr += v * ar[i]; }
    sl = wave_reduce_sum(sl);
    sr = wave_reduce_sum(sr);
    if (lane == 0) { gl[n] = sl; gr[n] = sr; }
}

// ---------------------------------------------------------------------------
// 6) edge logits (MX attention) + segment max    (one wave per edge)
// ---------------------------------------------------------------------------
__global__ void edge_logits_kernel(const int* __restrict__ src,
                                   const int* __restrict__ dst,
                                   const float* __restrict__ xp,
                                   const float* __restrict__ gl,
                                   const float* __restrict__ gr,
                                   float* __restrict__ eedge,
                                   unsigned* __restrict__ emax) {
    const int e    = (blockIdx.x * 256 + threadIdx.x) >> 5;   // exact E3 waves
    const int lane = threadIdx.x & 31;
    int s, d;
    if (e < EE) { s = src[e]; d = dst[e]; } else { s = d = e - EE; }
    const float* xi = xp + (size_t)d * FDIM + lane * 8;
    const float* xj = xp + (size_t)s * FDIM + lane * 8;
    float dot = 0.f;
#pragma unroll
    for (int i = 0; i < 8; ++i) dot += xi[i] * xj[i];
    dot = wave_reduce_sum(dot);
    if (lane == 0) {
        float logit = gl[s] + gr[d];
        float sig = 1.0f / (1.0f + expf(-dot));
        float v = logit * sig;
        v = (v >= 0.0f) ? v : NSLP * v;            // leaky relu
        eedge[e] = v;
        atomicMax(&emax[d], ford(v));
    }
}

// 7) a = exp(e - emax[d]); asum[d] += a      (thread per edge)
__global__ void edge_exp_kernel(const int* __restrict__ dst,
                                float* __restrict__ eedge,
                                const unsigned* __restrict__ emax,
                                float* __restrict__ asum) {
    const int e = blockIdx.x * 256 + threadIdx.x;
    if (e >= E3) return;
    const int d = (e < EE) ? dst[e] : e - EE;
    float a = expf(eedge[e] - dorf(emax[d]));
    eedge[e] = a;
    atomicAdd(&asum[d], a);
}

// 8) out[d] += xj * attn      (one wave per edge, 8 floats/lane)
__global__ void edge_scatter_kernel(const int* __restrict__ src,
                                    const int* __restrict__ dst,
                                    const float* __restrict__ xp,
                                    const float* __restrict__ eedge,
                                    const float* __restrict__ asum,
                                    float* __restrict__ out) {
    const int e    = (blockIdx.x * 256 + threadIdx.x) >> 5;   // exact E3 waves
    const int lane = threadIdx.x & 31;
    int s, d;
    if (e < EE) { s = src[e]; d = dst[e]; } else { s = d = e - EE; }
    const float attn = eedge[e] / (asum[d] + 1e-16f);
    const float* xj = xp  + (size_t)s * FDIM + lane * 8;
    float*       od = out + (size_t)d * FDIM + lane * 8;
#pragma unroll
    for (int i = 0; i < 8; ++i) atomicAdd(&od[i], xj[i] * attn);
}

// ---------------------------------------------------------------------------
// host launcher
// ---------------------------------------------------------------------------
extern "C" void kernel_launch(void* const* d_in, const int* in_sizes, int n_in,
                              void* d_out, int out_size, void* d_ws, size_t ws_size,
                              hipStream_t stream) {
    const float* x       = (const float*)d_in[0];
    const int*   eidx    = (const int*)  d_in[1];
    const float* sage_Wl = (const float*)d_in[2];
    const float* sage_bl = (const float*)d_in[3];
    const float* sage_Wr = (const float*)d_in[4];
    const float* gn_gam  = (const float*)d_in[5];
    const float* gn_bet  = (const float*)d_in[6];
    const float* gn_alp  = (const float*)d_in[7];
    const float* gc_Wrel = (const float*)d_in[8];
    const float* gc_brel = (const float*)d_in[9];
    const float* gc_Wrt  = (const float*)d_in[10];
    const float* sg_W    = (const float*)d_in[11];
    const float* sg_al   = (const float*)d_in[12];
    const float* sg_ar   = (const float*)d_in[13];
    const float* sg_bias = (const float*)d_in[14];
    float* out = (float*)d_out;

    const int* src = eidx;          // edge_index[0]
    const int* dst = eidx + EE;     // edge_index[1]

    // ---- workspace carve (keep every region 256B aligned) ----
    char* base = (char*)d_ws;
    size_t off = 0;
    auto carve = [&](size_t bytes) {
        char* p = base + off;
        off += (bytes + 255) & ~(size_t)255;
        return p;
    };
    float*    agg     = (float*)   carve((size_t)NN * FDIM * 4);   // f32 sums (atomics)
    float*    deg     = (float*)   carve((size_t)NN * 4);
    float*    invdeg  = (float*)   carve((size_t)NN * 4);
    float*    xc      = (float*)   carve((size_t)N2 * FDIM * 4);   // h1|h2 (pre-norm)
    float*    xp      = (float*)   carve((size_t)N2 * FDIM * 4);
    float*    gl      = (float*)   carve((size_t)N2 * 4);
    float*    gr      = (float*)   carve((size_t)N2 * 4);
    float*    stats   = (float*)   carve(4 * FDIM * 4);
    float*    ss      = (float*)   carve(4 * FDIM * 4);
    float*    eedge   = (float*)   carve((size_t)E3 * 4);
    unsigned* emax    = (unsigned*)carve((size_t)N2 * 4);
    float*    asum    = (float*)   carve((size_t)N2 * 4);
    __bf16*   x_bf    = (__bf16*)  carve((size_t)NN * FDIM * 2);
    __bf16*   agg_bf  = (__bf16*)  carve((size_t)NN * FDIM * 2);
    __bf16*   mean_bf = (__bf16*)  carve((size_t)NN * FDIM * 2);
    __bf16*   xcb     = (__bf16*)  carve((size_t)N2 * FDIM * 2);
    __bf16*   WlT     = (__bf16*)  carve((size_t)FDIM * FDIM * 2);
    __bf16*   WrT     = (__bf16*)  carve((size_t)FDIM * FDIM * 2);
    __bf16*   WrelT   = (__bf16*)  carve((size_t)FDIM * FDIM * 2);
    __bf16*   WrootT  = (__bf16*)  carve((size_t)FDIM * FDIM * 2);
    __bf16*   sgWT    = (__bf16*)  carve((size_t)FDIM * FDIM * 2);

    const int WBLK = (FDIM * FDIM) / 256;     // 256 blocks per weight pack

    // ---- init ----
    fillf_kernel<<<(NN * FDIM + 255) / 256, 256, 0, stream>>>(agg, NN * FDIM, 0.f);
    fillf_kernel<<<(NN + 255) / 256, 256, 0, stream>>>(deg, NN, 0.f);
    fillf_kernel<<<(4 * FDIM + 255) / 256, 256, 0, stream>>>(stats, 4 * FDIM, 0.f);
    fillf_kernel<<<(N2 + 255) / 256, 256, 0, stream>>>(asum, N2, 0.f);
    fillu_kernel<<<(N2 + 255) / 256, 256, 0, stream>>>(emax, N2, 0u);
    out_init_kernel<<<(N2 * FDIM) / 256, 256, 0, stream>>>(out, sg_bias);

    // ---- pack weights (once) and x ----
    packWt_kernel<<<WBLK, 256, 0, stream>>>(sage_Wl, WlT);
    packWt_kernel<<<WBLK, 256, 0, stream>>>(sage_Wr, WrT);
    packWt_kernel<<<WBLK, 256, 0, stream>>>(gc_Wrel, WrelT);
    packWt_kernel<<<WBLK, 256, 0, stream>>>(gc_Wrt, WrootT);
    packWt_kernel<<<WBLK, 256, 0, stream>>>(sg_W, sgWT);
    packbf_kernel<<<(NN * FDIM + 255) / 256, 256, 0, stream>>>(x, x_bf, NN * FDIM);

    // ---- 1: neighbor aggregation ----
    scatter_agg_kernel<<<EE / 8, 256, 0, stream>>>(src, dst, x, agg, deg);
    invdeg_kernel<<<(NN + 255) / 256, 256, 0, stream>>>(deg, invdeg);
    pack_agg_kernel<<<(NN * FDIM) / 256, 256, 0, stream>>>(agg, invdeg, agg_bf, mean_bf);

    // ---- 2: fused conv1/conv2 GEMMs (WMMA) : 50000 tiles / 8 waves = 6250 blocks
    gemm_h_kernel<<<6250, 256, 0, stream>>>(x_bf, agg_bf, mean_bf,
                                            WlT, WrT, WrelT, WrootT,
                                            sage_bl, gc_brel, xc);

    // ---- 3: GraphNorm + ReLU (emit bf16 activations) ----
    colstats_kernel<<<200, 256, 0, stream>>>(xc, stats, NN / 200);
    scaleshift_kernel<<<1, 256, 0, stream>>>(stats, gn_gam, gn_bet, gn_alp, ss);
    norm_relu_kernel<<<(N2 * FDIM) / 256, 256, 0, stream>>>(xc, ss, xcb);

    // ---- 4: xp = xcb @ sg_W (WMMA) : 100000 tiles / 8 waves = 12500 blocks
    gemm_xp_kernel<<<12500, 256, 0, stream>>>(xcb, sgWT, xp);

    // ---- 5-8: SuperGAT attention ----
    node_att_kernel<<<N2 / 8, 256, 0, stream>>>(xp, sg_al, sg_ar, gl, gr);
    edge_logits_kernel<<<E3 / 8, 256, 0, stream>>>(src, dst, xp, gl, gr, eedge, emax);
    edge_exp_kernel<<<(E3 + 255) / 256, 256, 0, stream>>>(dst, eedge, emax, asum);
    edge_scatter_kernel<<<E3 / 8, 256, 0, stream>>>(src, dst, xp, eedge, asum, out);
}
